// Recycling_50190987821738
// MI455X (gfx1250) — matrix operand order repaired and checked by reading, not compiled, not run on hardware
//
#include <hip/hip_runtime.h>
#include <hip/hip_bf16.h>

typedef __bf16 v16bf __attribute__((ext_vector_type(16)));
typedef float  v8f   __attribute__((ext_vector_type(8)));

#define L_SEQ 1024
#define D_PAIR 128
#define D_RBF_C 64
#define D_STATE_C 32
#define A_STRIDE 72   // 64 rbf + pad, multiple of 8 (16B-aligned chunks)
#define B_STRIDE 72

static __device__ __forceinline__ unsigned short f2bf(float f) {
    unsigned int u = __builtin_bit_cast(unsigned int, f);
    unsigned int r = u + 0x7FFFu + ((u >> 16) & 1u);
    return (unsigned short)(r >> 16);
}

static __device__ __forceinline__ void wred2(float& s, float& s2) {
    #pragma unroll
    for (int o = 16; o > 0; o >>= 1) {
        s  += __shfl_xor(s,  o, 32);
        s2 += __shfl_xor(s2, o, 32);
    }
}

// ---------------- msa layernorm: 1024 rows x 256 ----------------
__global__ __launch_bounds__(256) void msa_ln_kernel(
    const float* __restrict__ msa, const float* __restrict__ g,
    const float* __restrict__ be, float* __restrict__ out)
{
    const int row = blockIdx.x;
    const int tid = threadIdx.x;
    const int lane = tid & 31, wave = tid >> 5;
    float x = msa[row * 256 + tid];
    float s = x, s2 = x * x;
    wred2(s, s2);
    __shared__ float rs[8], rs2[8];
    if (lane == 0) { rs[wave] = s; rs2[wave] = s2; }
    __syncthreads();
    float tot = 0.f, tot2 = 0.f;
    #pragma unroll
    for (int w = 0; w < 8; ++w) { tot += rs[w]; tot2 += rs2[w]; }
    float mu = tot * (1.0f / 256.0f);
    float var = tot2 * (1.0f / 256.0f) - mu * mu;
    float rstd = rsqrtf(var + 1e-5f);
    out[row * 256 + tid] = (x - mu) * rstd * g[tid] + be[tid];
}

// ---------------- state layernorm: 1024 rows x 32 ----------------
__global__ __launch_bounds__(256) void state_ln_kernel(
    const float* __restrict__ st, const float* __restrict__ g,
    const float* __restrict__ be, float* __restrict__ out)
{
    const int lane = threadIdx.x & 31, wave = threadIdx.x >> 5;
    const int row = blockIdx.x * 8 + wave;
    float x = st[row * 32 + lane];
    float s = x, s2 = x * x;
    wred2(s, s2);
    float mu = s * (1.0f / 32.0f);
    float var = s2 * (1.0f / 32.0f) - mu * mu;
    float rstd = rsqrtf(var + 1e-5f);
    out[row * 32 + lane] = (x - mu) * rstd * g[lane] + be[lane];
}

// ---------------- fused pair kernel: one 16x16 (i,j) tile per block ----------------
__global__ __launch_bounds__(256) void pair_fused_kernel(
    const float* __restrict__ pair,  const float* __restrict__ state,
    const float* __restrict__ xyz,   const float* __restrict__ w_proj,
    const float* __restrict__ b_proj,const float* __restrict__ g_pair,
    const float* __restrict__ beta_pair,
    const float* __restrict__ g_state, const float* __restrict__ beta_state,
    float* __restrict__ out_pair)
{
    __shared__ unsigned short A_lds[256 * A_STRIDE];   // rbf, bf16 bits
    __shared__ unsigned short Bt_lds[128 * B_STRIDE];  // w_rbf^T rows, bf16 bits
    __shared__ float mu_lds[256], rstd_lds[256];
    __shared__ float sn_lds[32 * 32];                  // state_n: rows 0-15 = i, 16-31 = j
    __shared__ float cb_lds[32 * 4];                   // Cb points (xyz + pad)
    __shared__ float left_lds[16 * 128], right_lds[16 * 128];

    const int tid  = threadIdx.x;
    const int lane = tid & 31;
    const int wave = tid >> 5;
    const int i0 = blockIdx.y * 16;
    const int j0 = blockIdx.x * 16;

    // ---- stage 1: Cb + state_n for the 32 rows this tile touches ----
    if (tid < 32) {
        const int row = (tid < 16) ? (i0 + tid) : (j0 + (tid - 16));
        const float* xp = xyz + row * 9;
        float Nx = xp[0], Ny = xp[1], Nz = xp[2];
        float Ax = xp[3], Ay = xp[4], Az = xp[5];   // Ca
        float Cx = xp[6], Cy = xp[7], Cz = xp[8];
        float bx = Ax - Nx, by = Ay - Ny, bz = Az - Nz;
        float cx = Cx - Ax, cy = Cy - Ay, cz = Cz - Az;
        float ax = by * cz - bz * cy;
        float ay = bz * cx - bx * cz;
        float az = bx * cy - by * cx;
        cb_lds[tid * 4 + 0] = -0.58273431f * ax + 0.56802827f * bx - 0.54067466f * cx + Ax;
        cb_lds[tid * 4 + 1] = -0.58273431f * ay + 0.56802827f * by - 0.54067466f * cy + Ay;
        cb_lds[tid * 4 + 2] = -0.58273431f * az + 0.56802827f * bz - 0.54067466f * cz + Az;

        const float* sp = state + row * 32;
        float s = 0.f, s2 = 0.f;
        for (int k = 0; k < 32; ++k) { float v = sp[k]; s += v; s2 += v * v; }
        float mu = s * (1.0f / 32.0f);
        float var = s2 * (1.0f / 32.0f) - mu * mu;
        float rstd = rsqrtf(var + 1e-5f);
        for (int k = 0; k < 32; ++k)
            sn_lds[tid * 32 + k] = (sp[k] - mu) * rstd * g_state[k] + beta_state[k];
    }
    __syncthreads();

    // ---- stage 2a: RBF -> A matrix (M=256 pairs, K=64), bf16 in LDS ----
    {
        const int iL = tid >> 4, jL = tid & 15;
        float dx = cb_lds[iL * 4 + 0] - cb_lds[(16 + jL) * 4 + 0];
        float dy = cb_lds[iL * 4 + 1] - cb_lds[(16 + jL) * 4 + 1];
        float dz = cb_lds[iL * 4 + 2] - cb_lds[(16 + jL) * 4 + 2];
        float D = sqrtf(fmaxf(dx * dx + dy * dy + dz * dz, 1e-12f));
        unsigned short* arow = A_lds + tid * A_STRIDE;
        for (int r = 0; r < D_RBF_C; ++r) {
            float m = 2.0f + (20.0f / 63.0f) * (float)r;
            float t = (D - m) * 3.2f;                 // 1/sigma = 64/20
            arow[r] = f2bf(__expf(-t * t));
        }
    }
    // ---- stage 2b: B^T = w_proj[:, :64] rows, bf16 in LDS ----
    {
        const int p = tid >> 1;
        const int c0 = (tid & 1) * 32;
        for (int c = 0; c < 32; ++c)
            Bt_lds[p * B_STRIDE + c0 + c] = f2bf(w_proj[p * 128 + c0 + c]);
    }
    // ---- stage 2c: left_p / right_p for this tile ----
    for (int idx = tid; idx < 2048; idx += 256) {
        const int li = idx >> 7, p = idx & 127;
        const float* wl = w_proj + p * 128 + 64;
        const float* wr = w_proj + p * 128 + 96;
        float al = 0.f, ar = 0.f;
        for (int s = 0; s < 32; ++s) {
            al += sn_lds[li * 32 + s] * wl[s];
            ar += sn_lds[(16 + li) * 32 + s] * wr[s];
        }
        left_lds[li * 128 + p] = al;
        right_lds[li * 128 + p] = ar;
    }
    // ---- stage 2d: per-pair mean/rstd over 128 channels (wave handles 32 rows) ----
    for (int k = 0; k < 32; ++k) {
        const int m = wave * 32 + k;
        const int gi = i0 + (m >> 4), gj = j0 + (m & 15);
        const float4* pr = (const float4*)(pair + ((size_t)gi * L_SEQ + gj) * D_PAIR);
        float4 v = pr[lane];
        float s  = v.x + v.y + v.z + v.w;
        float s2 = v.x * v.x + v.y * v.y + v.z * v.z + v.w * v.w;
        wred2(s, s2);
        if (lane == 0) {
            float mu = s * (1.0f / 128.0f);
            mu_lds[m] = mu;
            rstd_lds[m] = rsqrtf(s2 * (1.0f / 128.0f) - mu * mu + 1e-5f);
        }
    }
    __syncthreads();

    // ---- stage 3: WMMA edge GEMM + fused layernorm/add/store ----
    const int n = lane & 15;
    const int half = lane >> 4;
    union Frag { uint4 q[2]; v16bf v; };

    for (int t = wave; t < 128; t += 8) {
        const int mt = t >> 3;        // which group of 16 pairs (fixed i = i0+mt)
        const int nt = t & 7;         // which group of 16 channels
        const int p = nt * 16 + n;
        const int gi = i0 + mt;

        // A fragments (16x32 bf16): lane row M = lane&15; e<8 -> K=8*half+e, e>=8 -> K=16+8*half+(e-8)
        const unsigned short* arow = A_lds + (mt * 16 + n) * A_STRIDE;
        Frag a0, a1, b0, b1;
        a0.q[0] = *(const uint4*)(arow + 8 * half);
        a0.q[1] = *(const uint4*)(arow + 8 * half + 16);
        a1.q[0] = *(const uint4*)(arow + 32 + 8 * half);
        a1.q[1] = *(const uint4*)(arow + 32 + 8 * half + 16);
        // B fragments (32x16 bf16): lane col N = lane&15; e<8 -> K=8*half+e, e>=8 -> K=16+8*half+(e-8)
        const unsigned short* brow = Bt_lds + p * B_STRIDE;
        b0.q[0] = *(const uint4*)(brow + 8 * half);
        b0.q[1] = *(const uint4*)(brow + 8 * half + 16);
        b1.q[0] = *(const uint4*)(brow + 32 + 8 * half);
        b1.q[1] = *(const uint4*)(brow + 32 + 8 * half + 16);

        v8f acc = {0.f, 0.f, 0.f, 0.f, 0.f, 0.f, 0.f, 0.f};
        acc = __builtin_amdgcn_wmma_f32_16x16x32_bf16(false, a0.v, false, b0.v,
                                                      (short)0, acc, false, false);
        acc = __builtin_amdgcn_wmma_f32_16x16x32_bf16(false, a1.v, false, b1.v,
                                                      (short)0, acc, false, false);

        const float lp = left_lds[mt * 128 + p];
        const float gp = g_pair[p], bp = beta_pair[p], bb = b_proj[p];
        const size_t base = ((size_t)gi * L_SEQ + j0) * D_PAIR + p;

        #pragma unroll
        for (int r = 0; r < 8; ++r) {
            const int jl = r + half * 8;             // D layout: M = r + 8*half
            const int m = mt * 16 + jl;
            const size_t off = base + (size_t)jl * D_PAIR;
            float pv = pair[off];                    // L2/WGP$ hit (tile just read)
            float o = (pv - mu_lds[m]) * rstd_lds[m] * gp + bp
                      + acc[r] + lp + right_lds[jl * 128 + p] + bb;
            out_pair[off] = o;
        }
    }
}

extern "C" void kernel_launch(void* const* d_in, const int* in_sizes, int n_in,
                              void* d_out, int out_size, void* d_ws, size_t ws_size,
                              hipStream_t stream) {
    (void)in_sizes; (void)n_in; (void)out_size; (void)d_ws; (void)ws_size;
    const float* msa        = (const float*)d_in[1];
    const float* pair       = (const float*)d_in[2];
    const float* state      = (const float*)d_in[3];
    const float* xyz        = (const float*)d_in[4];
    const float* w_proj     = (const float*)d_in[6];
    const float* b_proj     = (const float*)d_in[7];
    const float* g_pair     = (const float*)d_in[8];
    const float* beta_pair  = (const float*)d_in[9];
    const float* g_msa      = (const float*)d_in[10];
    const float* beta_msa   = (const float*)d_in[11];
    const float* g_state    = (const float*)d_in[12];
    const float* beta_state = (const float*)d_in[13];

    float* out       = (float*)d_out;
    float* out_msa   = out;                        // 1024*256
    float* out_pair  = out + 262144;               // 1024*1024*128
    float* out_state = out + 134479872;            // 1024*32

    msa_ln_kernel<<<1024, 256, 0, stream>>>(msa, g_msa, beta_msa, out_msa);
    state_ln_kernel<<<128, 256, 0, stream>>>(state, g_state, beta_state, out_state);
    dim3 grid(64, 64);
    pair_fused_kernel<<<grid, 256, 0, stream>>>(pair, state, xyz, w_proj, b_proj,
                                                g_pair, beta_pair, g_state, beta_state,
                                                out_pair);
}
